// Attention_65180423685043
// MI455X (gfx1250) — matrix operand order repaired and checked by reading, not compiled, not run on hardware
//
#include <hip/hip_runtime.h>
#include <hip/hip_bf16.h>

#define B_     4
#define N_     4096
#define DIM_   256
#define HEADS_ 4
#define HD_    64
#define M_     1024
#define HS_    32
#define WSP_   32
#define HP_    64
#define WP_    64
#define SCALE_ 0.125f

typedef __attribute__((ext_vector_type(16))) __bf16 v16bf;
typedef __attribute__((ext_vector_type(8)))  float  v8f;
typedef __attribute__((ext_vector_type(4)))  unsigned int u32x4;
typedef __attribute__((ext_vector_type(8)))  int i32x8;
typedef __attribute__((ext_vector_type(4)))  int i32x4;

#if defined(__has_builtin)
#if __has_builtin(__builtin_amdgcn_tensor_load_to_lds) && __has_builtin(__builtin_amdgcn_s_wait_tensorcnt)
#define HAVE_TDM 1
#endif
#endif

// ---------------- Tensor Data Mover: 1-D contiguous tile (bf16 elements) global -> LDS ---------
#if HAVE_TDM
__device__ __forceinline__ void tdm_load_1d(unsigned lds_off, const void* gptr, unsigned nelem2B) {
  unsigned long long ga = (unsigned long long)(uintptr_t)gptr;
  u32x4 g0;
  g0[0] = 1u;                                              // count=1, user descriptor
  g0[1] = lds_off;                                         // lds_addr
  g0[2] = (unsigned)ga;                                    // global_addr[31:0]
  g0[3] = (unsigned)((ga >> 32) & 0x01FFFFFFull) | (2u << 30);  // addr[56:32] | type=2
  i32x8 g1;
  g1[0] = (int)(1u << 16);                                 // data_size=1 (2 bytes)
  g1[1] = (int)((nelem2B & 0xFFFFu) << 16);                // tensor_dim0[15:0]
  g1[2] = (int)(((nelem2B >> 16) & 0xFFFFu) | (1u << 16)); // tensor_dim0[31:16] | tensor_dim1=1
  g1[3] = (int)((nelem2B & 0xFFFFu) << 16);                // tile_dim0
  g1[4] = 1;                                               // tile_dim1 = 1
  g1[5] = (int)nelem2B;                                    // tensor_dim0_stride lo
  g1[6] = 0;
  g1[7] = 0;
  i32x4 z = {0, 0, 0, 0};
#if __clang_major__ >= 23
  i32x8 z8 = {};
  __builtin_amdgcn_tensor_load_to_lds(g0, g1, z, z, z8, 0);
#else
  __builtin_amdgcn_tensor_load_to_lds(g0, g1, z, z, 0);
#endif
}
#endif

// ---------------- WMMA fragment builders (wave32, contiguous k = 16*(lane/16) + s in A and B) ----------------
__device__ __forceinline__ v16bf a_frag_bf16(const __bf16* A, int ld, int lane) {
  int r = lane & 15, kh = (lane >> 4) << 4;
  const __bf16* p = A + r * ld;
  v16bf a;
#pragma unroll
  for (int j = 0; j < 4; ++j) {
    a[2*j]     = p[kh + 2*j];
    a[2*j + 1] = p[kh + 2*j + 1];
    a[8 + 2*j]     = p[8 + kh + 2*j];
    a[8 + 2*j + 1] = p[8 + kh + 2*j + 1];
  }
  return a;
}
// B 32x16 where element(k,n) = src[n*ld + k]  (B^T stored row-major by n)
__device__ __forceinline__ v16bf bt_frag_bf16(const __bf16* Bw, int ld, int lane) {
  int n = lane & 15, kb = (lane >> 4) << 4;
  const __bf16* p = Bw + n * ld + kb;
  v16bf b;
#pragma unroll
  for (int j = 0; j < 16; ++j) b[j] = p[j];
  return b;
}
// B 32x16 where element(k,n) = src[k*ld + n]  (row-major B, e.g. V tile)
__device__ __forceinline__ v16bf b_frag_rm_bf16(const __bf16* Bm, int ld, int lane) {
  int n = lane & 15, kb = (lane >> 4) << 4;
  v16bf b;
#pragma unroll
  for (int j = 0; j < 16; ++j) b[j] = Bm[(kb + j) * ld + n];
  return b;
}

#define WMMA_BF16(A, Bv, C) \
  __builtin_amdgcn_wmma_f32_16x16x32_bf16(false, (A), false, (Bv), (short)0, (C), false, false)

__device__ __forceinline__ float hr_max(float x) {
#pragma unroll
  for (int m = 1; m <= 8; m <<= 1) x = fmaxf(x, __shfl_xor(x, m, 32));
  return x;
}
__device__ __forceinline__ float hr_sum(float x) {
#pragma unroll
  for (int m = 1; m <= 8; m <<= 1) x += __shfl_xor(x, m, 32);
  return x;
}

// ---------------- Kernel 0: fp32 -> bf16 convert ----------------
__global__ void __launch_bounds__(256) cvt_bf16_kernel(const float* __restrict__ in,
                                                       __bf16* __restrict__ out, int n) {
  int i = blockIdx.x * 256 + threadIdx.x;
  if (i < n) out[i] = (__bf16)in[i];
}

// ---------------- Generic bf16 WMMA GEMM body: C[16x64 per wave] = A @ B^T ----------------
// All 8 waves of a block share one 64-column group; B tile staged in LDS (32 KB).
// B fragments for all 4 tiles are preloaded before the WMMA chain so the scheduler can
// issue 8 back-to-back ds_load_b128 and stagger the dscnt waits across the 4 WMMAs.
#define GEMM_BODY(Aglob, Bglob, MT)                                                    \
  __shared__ __bf16 sB[64 * DIM_];                                                     \
  const int lane = threadIdx.x & 31;                                                   \
  const int wid  = blockIdx.x * 8 + (threadIdx.x >> 5);                                \
  const int mTile = wid % (MT);                                                        \
  const int nGrp  = wid / (MT);                                                        \
  {                                                                                    \
    const uint4* srcv = (const uint4*)((Bglob) + (size_t)nGrp * 64 * DIM_);            \
    uint4* dstv = (uint4*)sB;                                                          \
    for (int i = threadIdx.x; i < (64 * DIM_) / 8; i += 256) dstv[i] = srcv[i];        \
  }                                                                                    \
  __syncthreads();                                                                     \
  const __bf16* Abase = (Aglob) + (size_t)mTile * 16 * DIM_;                           \
  v8f acc[4] = {};                                                                     \
  for (int k0 = 0; k0 < DIM_; k0 += 32) {                                              \
    __builtin_prefetch(Abase + k0 + 64, 0, 0);                                         \
    v16bf a = a_frag_bf16(Abase + k0, DIM_, lane);                                     \
    v16bf btf[4];                                                                      \
    _Pragma("unroll")                                                                  \
    for (int t = 0; t < 4; ++t)                                                        \
      btf[t] = bt_frag_bf16(sB + (size_t)(t * 16) * DIM_ + k0, DIM_, lane);            \
    _Pragma("unroll")                                                                  \
    for (int t = 0; t < 4; ++t)                                                        \
      acc[t] = WMMA_BF16(a, btf[t], acc[t]);                                           \
  }                                                                                    \
  const int hh = lane >> 4, cl = lane & 15;

// ---------------- Kernel 1: Q projection -> bf16 (B,H,N,hd) ----------------
__global__ void __launch_bounds__(256) qproj_kernel(const __bf16* __restrict__ xb,
                                                    const __bf16* __restrict__ qwb,
                                                    const float* __restrict__ qbias,
                                                    __bf16* __restrict__ qout) {
  GEMM_BODY(xb, qwb, (B_ * N_) / 16)
#pragma unroll
  for (int t = 0; t < 4; ++t)
#pragma unroll
    for (int r = 0; r < 8; ++r) {
      int row = mTile * 16 + r + 8 * hh;
      int col = nGrp * 64 + t * 16 + cl;
      float v = acc[t][r] + qbias[col];
      int b = row >> 12, n = row & (N_ - 1);
      int h = col >> 6, d = col & 63;
      qout[(((size_t)(b * HEADS_ + h) * N_) + n) * HD_ + d] = (__bf16)v;
    }
}

// ---------------- Kernel 2: SR depthwise conv 3x3 s2 + bias + LayerNorm -> x_kv bf16 ----------
__global__ void __launch_bounds__(256) srconv_ln_kernel(const float* __restrict__ x,
                                                        const float* __restrict__ srw,
                                                        const float* __restrict__ srb,
                                                        const float* __restrict__ srg,
                                                        const float* __restrict__ srbe,
                                                        __bf16* __restrict__ xkvb) {
  __shared__ float sred[DIM_];
  const int bm = blockIdx.x;
  const int b = bm / M_, m = bm % M_;
  const int hs = m / WSP_, ws = m % WSP_;
  const int c = threadIdx.x;
  float acc = srb[c];
#pragma unroll
  for (int dy = 0; dy < 3; ++dy)
#pragma unroll
    for (int dx = 0; dx < 3; ++dx) {
      int y = hs * 2 + dy - 1, xx = ws * 2 + dx - 1;
      if (y >= 0 && y < HP_ && xx >= 0 && xx < WP_)
        acc += x[((size_t)b * N_ + y * WP_ + xx) * DIM_ + c] * srw[c * 9 + dy * 3 + dx];
    }
  sred[c] = acc;  __syncthreads();
  for (int s = 128; s > 0; s >>= 1) { if (c < s) sred[c] += sred[c + s]; __syncthreads(); }
  float mean = sred[0] * (1.0f / DIM_);
  __syncthreads();
  float dv = acc - mean;
  sred[c] = dv * dv;  __syncthreads();
  for (int s = 128; s > 0; s >>= 1) { if (c < s) sred[c] += sred[c + s]; __syncthreads(); }
  float var = sred[0] * (1.0f / DIM_);
  xkvb[((size_t)b * M_ + m) * DIM_ + c] = (__bf16)(dv * rsqrtf(var + 1e-6f) * srg[c] + srbe[c]);
}

// ---------------- Kernel 3: KV projection ----------------
__global__ void __launch_bounds__(256) kvproj_kernel(const __bf16* __restrict__ xkvb,
                                                     const __bf16* __restrict__ kvwb,
                                                     const float* __restrict__ kvb,
                                                     __bf16* __restrict__ kout,
                                                     __bf16* __restrict__ vout,
                                                     float* __restrict__ vf) {
  GEMM_BODY(xkvb, kvwb, (B_ * M_) / 16)
#pragma unroll
  for (int t = 0; t < 4; ++t)
#pragma unroll
    for (int r = 0; r < 8; ++r) {
      int row = mTile * 16 + r + 8 * hh;
      int col = nGrp * 64 + t * 16 + cl;      // 0..511
      float v = acc[t][r] + kvb[col];
      int b = row / M_, m = row % M_;
      int kvsel = col >> 8, rem = col & 255;
      int h = rem >> 6, d = rem & 63;
      size_t idx = (((size_t)(b * HEADS_ + h) * M_) + m) * HD_ + d;
      if (kvsel == 0) kout[idx] = (__bf16)v;
      else { vout[idx] = (__bf16)v; vf[idx] = v; }
    }
}

// ---------------- Kernel 4: flash attention (double-buffered TDM K/V tiles) ----------------
__global__ void __launch_bounds__(128) attn_kernel(const __bf16* __restrict__ qg,
                                                   const __bf16* __restrict__ kg,
                                                   const __bf16* __restrict__ vg,
                                                   float* __restrict__ og) {
  __shared__ __bf16 sK[2][32 * HD_];
  __shared__ __bf16 sV[2][32 * HD_];
  __shared__ __bf16 sP[4][16 * 32];
  const int lane = threadIdx.x & 31;
  const int wave = threadIdx.x >> 5;
  const int bh   = blockIdx.y;
  const int bidx = bh >> 2, hidx = bh & 3;
  const int qRow0 = blockIdx.x * 64 + wave * 16;
  const __bf16* qbase = qg + ((size_t)bh * N_ + qRow0) * HD_;
  v16bf qa0 = a_frag_bf16(qbase, HD_, lane);
  v16bf qa1 = a_frag_bf16(qbase + 32, HD_, lane);
  const __bf16* kbase = kg + (size_t)bh * M_ * HD_;
  const __bf16* vbase = vg + (size_t)bh * M_ * HD_;
  float mstate[8], lstate[8];
  v8f oacc[4] = {};
#pragma unroll
  for (int r = 0; r < 8; ++r) { mstate[r] = -1e30f; lstate[r] = 0.0f; }
  const int hh = lane >> 4, cl = lane & 15;

#if HAVE_TDM
  // prologue: DMA first K/V tile into buffer 0
  if (wave == 0) {
    tdm_load_1d((unsigned)(uintptr_t)&sK[0][0], kbase, 32 * HD_);
    tdm_load_1d((unsigned)(uintptr_t)&sV[0][0], vbase, 32 * HD_);
  }
#endif

  for (int j0 = 0; j0 < M_; j0 += 32) {
    const int cur = (j0 >> 5) & 1;
    __syncthreads();   // all waves done reading the buffer the next DMA will overwrite
#if HAVE_TDM
    if (wave == 0) {
      if (j0 + 32 < M_) {
        // prefetch next tile into the other buffer; only require current tile's
        // two loads (issued earlier) to be complete -> DMA overlaps compute
        tdm_load_1d((unsigned)(uintptr_t)&sK[1 - cur][0], kbase + (size_t)(j0 + 32) * HD_, 32 * HD_);
        tdm_load_1d((unsigned)(uintptr_t)&sV[1 - cur][0], vbase + (size_t)(j0 + 32) * HD_, 32 * HD_);
        __builtin_amdgcn_s_wait_tensorcnt(2);
      } else {
        __builtin_amdgcn_s_wait_tensorcnt(0);
      }
    }
#else
    for (int i = threadIdx.x; i < 32 * HD_; i += 128) {
      sK[cur][i] = kbase[(size_t)j0 * HD_ + i];
      sV[cur][i] = vbase[(size_t)j0 * HD_ + i];
    }
#endif
    __syncthreads();   // current tile visible to all waves
    const __bf16* kt = &sK[cur][0];
    const __bf16* vt = &sV[cur][0];
    // S tiles: keys [0..15] and [16..31]
    v8f s0 = {}, s1 = {};
    s0 = WMMA_BF16(qa0, bt_frag_bf16(kt, HD_, lane), s0);
    s0 = WMMA_BF16(qa1, bt_frag_bf16(kt + 32, HD_, lane), s0);
    s1 = WMMA_BF16(qa0, bt_frag_bf16(kt + 16 * HD_, HD_, lane), s1);
    s1 = WMMA_BF16(qa1, bt_frag_bf16(kt + 16 * HD_ + 32, HD_, lane), s1);
    // online softmax (row stats per (reg, lane-half))
#pragma unroll
    for (int r = 0; r < 8; ++r) {
      float a0 = s0[r] * SCALE_, a1 = s1[r] * SCALE_;
      float mx   = hr_max(fmaxf(a0, a1));
      float mnew = fmaxf(mstate[r], mx);
      float corr = __expf(mstate[r] - mnew);
      float e0 = __expf(a0 - mnew);
      float e1 = __expf(a1 - mnew);
      float rs = hr_sum(e0 + e1);
      lstate[r] = lstate[r] * corr + rs;
      mstate[r] = mnew;
#pragma unroll
      for (int t = 0; t < 4; ++t) oacc[t][r] *= corr;
      int row = r + 8 * hh;
      sP[wave][row * 32 + cl]      = (__bf16)e0;
      sP[wave][row * 32 + 16 + cl] = (__bf16)e1;
    }
    // O += P @ V (sP is per-wave; wave-internal ds dependency handled by dscnt waits)
    v16bf pa = a_frag_bf16(&sP[wave][0], 32, lane);
    v16bf vbf[4];
#pragma unroll
    for (int t = 0; t < 4; ++t) vbf[t] = b_frag_rm_bf16(vt + t * 16, HD_, lane);
#pragma unroll
    for (int t = 0; t < 4; ++t) oacc[t] = WMMA_BF16(pa, vbf[t], oacc[t]);
  }
#pragma unroll
  for (int r = 0; r < 8; ++r) {
    float inv = 1.0f / lstate[r];
    int row = qRow0 + r + 8 * hh;
#pragma unroll
    for (int t = 0; t < 4; ++t)
      og[((size_t)(bidx * N_ + row)) * DIM_ + hidx * HD_ + t * 16 + cl] = oacc[t][r] * inv;
  }
}

// ---------------- Kernel 5: up conv + PixelShuffle + LN, fuse (o + LN(u)) -> bf16 A -----------
__global__ void __launch_bounds__(256) up_ln_kernel(const float* __restrict__ vf,
                                                    const float* __restrict__ upw,
                                                    const float* __restrict__ upb,
                                                    const float* __restrict__ upg,
                                                    const float* __restrict__ upbe,
                                                    const float* __restrict__ obuf,
                                                    __bf16* __restrict__ abuf) {
  __shared__ float sred[DIM_];
  const int bn = blockIdx.x;
  const int b = bn / N_, n = bn % N_;
  const int hp = n / WP_, wp = n % WP_;
  const int hs = hp >> 1, sy = hp & 1, ws = wp >> 1, sx = wp & 1;
  const int c = threadIdx.x;
  const int oc = c * 4 + sy * 2 + sx;
  const int h = c >> 6, d = c & 63;
  float acc = upb[oc];
#pragma unroll
  for (int dy = 0; dy < 3; ++dy)
#pragma unroll
    for (int dx = 0; dx < 3; ++dx) {
      int y = hs + dy - 1, xx = ws + dx - 1;
      if (y >= 0 && y < HS_ && xx >= 0 && xx < WSP_)
        acc += vf[(((size_t)(b * HEADS_ + h) * M_) + y * WSP_ + xx) * HD_ + d] *
               upw[oc * 9 + dy * 3 + dx];
    }
  sred[c] = acc;  __syncthreads();
  for (int s = 128; s > 0; s >>= 1) { if (c < s) sred[c] += sred[c + s]; __syncthreads(); }
  float mean = sred[0] * (1.0f / DIM_);
  __syncthreads();
  float dv = acc - mean;
  sred[c] = dv * dv;  __syncthreads();
  for (int s = 128; s > 0; s >>= 1) { if (c < s) sred[c] += sred[c + s]; __syncthreads(); }
  float var = sred[0] * (1.0f / DIM_);
  float u_ln = dv * rsqrtf(var + 1e-6f) * upg[c] + upbe[c];
  abuf[((size_t)b * N_ + n) * DIM_ + c] = (__bf16)(obuf[((size_t)b * N_ + n) * DIM_ + c] + u_ln);
}

// ---------------- Kernel 6: output projection -> fp32 out ----------------
__global__ void __launch_bounds__(256) proj_kernel(const __bf16* __restrict__ abuf,
                                                   const __bf16* __restrict__ pwb,
                                                   const float* __restrict__ pb,
                                                   float* __restrict__ out) {
  GEMM_BODY(abuf, pwb, (B_ * N_) / 16)
#pragma unroll
  for (int t = 0; t < 4; ++t)
#pragma unroll
    for (int r = 0; r < 8; ++r) {
      int row = mTile * 16 + r + 8 * hh;
      int col = nGrp * 64 + t * 16 + cl;
      out[(size_t)row * DIM_ + col] = acc[t][r] + pb[col];
    }
}

// ---------------- launch ----------------
extern "C" void kernel_launch(void* const* d_in, const int* in_sizes, int n_in,
                              void* d_out, int out_size, void* d_ws, size_t ws_size,
                              hipStream_t stream) {
  const float* x      = (const float*)d_in[0];
  const float* q_w    = (const float*)d_in[1];
  const float* q_b    = (const float*)d_in[2];
  const float* kv_w   = (const float*)d_in[3];
  const float* kv_b   = (const float*)d_in[4];
  const float* sr_w   = (const float*)d_in[5];
  const float* sr_b   = (const float*)d_in[6];
  const float* sr_g   = (const float*)d_in[7];
  const float* sr_be  = (const float*)d_in[8];
  const float* up_w   = (const float*)d_in[9];
  const float* up_b   = (const float*)d_in[10];
  const float* upn_g  = (const float*)d_in[11];
  const float* upn_be = (const float*)d_in[12];
  const float* proj_w = (const float*)d_in[13];
  const float* proj_b = (const float*)d_in[14];
  float* out = (float*)d_out;

  char* ws = (char*)d_ws;
  size_t off = 0;
  auto alloc = [&](size_t bytes) { void* p = ws + off; off = (off + bytes + 255) & ~(size_t)255; return p; };
  __bf16* xb16   = (__bf16*)alloc((size_t)B_ * N_ * DIM_ * 2);           // 8 MB
  __bf16* qwb    = (__bf16*)alloc((size_t)DIM_ * DIM_ * 2);              // 128 KB
  __bf16* kvwb   = (__bf16*)alloc((size_t)2 * DIM_ * DIM_ * 2);          // 256 KB
  __bf16* pwb    = (__bf16*)alloc((size_t)DIM_ * DIM_ * 2);              // 128 KB
  __bf16* qb_buf = (__bf16*)alloc((size_t)B_ * HEADS_ * N_ * HD_ * 2);   // 8 MB
  __bf16* kb_buf = (__bf16*)alloc((size_t)B_ * HEADS_ * M_ * HD_ * 2);   // 2 MB
  __bf16* vb_buf = (__bf16*)alloc((size_t)B_ * HEADS_ * M_ * HD_ * 2);   // 2 MB
  float*  vf_buf = (float*) alloc((size_t)B_ * HEADS_ * M_ * HD_ * 4);   // 4 MB
  __bf16* xkvb   = (__bf16*)alloc((size_t)B_ * M_ * DIM_ * 2);           // 2 MB
  float*  obuf   = (float*) alloc((size_t)B_ * N_ * DIM_ * 4);           // 16 MB
  __bf16* abuf   = (__bf16*)alloc((size_t)B_ * N_ * DIM_ * 2);           // 8 MB

  // 0. one-time fp32 -> bf16 conversions (activations + GEMM weights)
  cvt_bf16_kernel<<<(B_ * N_ * DIM_ + 255) / 256, 256, 0, stream>>>(x, xb16, B_ * N_ * DIM_);
  cvt_bf16_kernel<<<(DIM_ * DIM_ + 255) / 256, 256, 0, stream>>>(q_w, qwb, DIM_ * DIM_);
  cvt_bf16_kernel<<<(2 * DIM_ * DIM_ + 255) / 256, 256, 0, stream>>>(kv_w, kvwb, 2 * DIM_ * DIM_);
  cvt_bf16_kernel<<<(DIM_ * DIM_ + 255) / 256, 256, 0, stream>>>(proj_w, pwb, DIM_ * DIM_);
  // 1. Q projection (GEMM 16384x256x256, LDS-staged bf16 weights)
  qproj_kernel<<<512, 256, 0, stream>>>(xb16, qwb, q_b, qb_buf);
  // 2. spatial-reduction depthwise conv + LN -> bf16
  srconv_ln_kernel<<<B_ * M_, DIM_, 0, stream>>>(x, sr_w, sr_b, sr_g, sr_be, xkvb);
  // 3. KV projection (GEMM 4096x512x256)
  kvproj_kernel<<<256, 256, 0, stream>>>(xkvb, kvwb, kv_b, kb_buf, vb_buf, vf_buf);
  // 4. flash attention (double-buffered TDM K/V staging + WMMA QK^T / PV)
  attn_kernel<<<dim3(N_ / 64, B_ * HEADS_), 128, 0, stream>>>(qb_buf, kb_buf, vb_buf, obuf);
  // 5. up-branch depthwise conv + PixelShuffle + LN, fused residual add -> bf16 A
  up_ln_kernel<<<B_ * N_, DIM_, 0, stream>>>(vf_buf, up_w, up_b, upn_g, upn_be, obuf, abuf);
  // 6. output projection (GEMM 16384x256x256) -> fp32 out
  proj_kernel<<<512, 256, 0, stream>>>(abuf, pwb, proj_b, out);
}
